// GCNEncoder_32186484916266
// MI455X (gfx1250) — matrix operand order repaired and checked by reading, not compiled, not run on hardware
//
#include <hip/hip_runtime.h>

// ---------------------------------------------------------------------------
// GCN 2-layer forward for MI455X (gfx1250, wave32, WMMA).
//   h = relu( Âx W1 + b1 );  out = log_softmax( Â h W2 + b2 )
//   Â = D^-1/2 (A+I) D^-1/2
// GEMMs: v_wmma_f32_16x16x32_bf16 (bf16 in, f32 accumulate), register-blocked
// over the full N so A is streamed from HBM exactly once.
// ---------------------------------------------------------------------------

typedef __attribute__((ext_vector_type(16))) __bf16 v16bf;
typedef __attribute__((ext_vector_type(8)))  __bf16 v8bf;
typedef __attribute__((ext_vector_type(8)))  float  v8f;

#define N_NODES 100000
#define N_EDGES 1600000
#define IN_CH   500
#define KPAD1   512     // IN_CH padded to multiple of 32 for WMMA K
#define HIDDEN  128
#define OUT_CH  64

// ---------------------------------------------------------------------------
// degree / normalization
// ---------------------------------------------------------------------------
__global__ void k_fill(float* p, float v, int n) {
    int i = blockIdx.x * 256 + threadIdx.x;
    if (i < n) p[i] = v;
}

__global__ void k_deg(const int* __restrict__ ei, float* __restrict__ deg, int nE, int nN) {
    int e = blockIdx.x * 256 + threadIdx.x;
    if (e >= nE) return;
    int d = ei[nE + e];                 // dst row of edge_index
    if ((unsigned)d < (unsigned)nN) atomicAdd(&deg[d], 1.0f);
}

__global__ void k_rsqrt_inplace(float* p, int n) {
    int i = blockIdx.x * 256 + threadIdx.x;
    if (i < n) p[i] = rsqrtf(fmaxf(p[i], 1.0f));
}

// ---------------------------------------------------------------------------
// f32 -> bf16 conversions (pad inner / outer dims with zeros)
// ---------------------------------------------------------------------------
__global__ void k_cvt_pad_cols(const float* __restrict__ src, __bf16* __restrict__ dst,
                               int cin, int cpad) {
    int c = blockIdx.x * 256 + threadIdx.x;
    long r = blockIdx.y;
    if (c >= cpad) return;
    float v = (c < cin) ? src[r * (long)cin + c] : 0.0f;
    dst[r * (long)cpad + c] = (__bf16)v;
}

__global__ void k_cvt_pad_rows(const float* __restrict__ src, __bf16* __restrict__ dst,
                               int rin, int rpad, int cols) {
    int i = blockIdx.x * 256 + threadIdx.x;
    if (i >= rpad * cols) return;
    int r = i / cols;
    float v = (r < rin) ? src[i] : 0.0f;
    dst[i] = (__bf16)v;
}

// ---------------------------------------------------------------------------
// WMMA GEMM:  C[M,N] = A[M,K](bf16) * W[K,N](bf16), f32 accumulate.
// Templated on N (power of two, <=128). 256 threads = 8 wave32s; each wave
// owns one 16-row M-tile and ALL N/16 tiles of it (N/16 v8f accumulators in
// registers), so each A fragment feeds N/16 WMMAs and A is read from global
// exactly once. W is staged per 128-deep K chunk, all N columns, in LDS laid
// out [n][k] with a +8 bf16 row pad (stride 136 elems = 68 dwords -> lanes
// rotate 4 banks/row: conflict-free b128 DS reads).
// Requires: M % 16 == 0, K % 128 == 0.
// ---------------------------------------------------------------------------
#define KC 128
#define LDSW_STRIDE (KC + 8)   // 136 bf16

template <int N>
__global__ __launch_bounds__(256) void k_gemm_wmma(
        const __bf16* __restrict__ A, const __bf16* __restrict__ W,
        float* __restrict__ C, int M, int K) {
    constexpr int NT     = N / 16;                 // N-tiles per wave
    constexpr int NSHIFT = (N == 128) ? 7 : 6;
    __shared__ __align__(32) __bf16 ldsW[N * LDSW_STRIDE];

    const int wave  = threadIdx.x >> 5;            // 0..7
    const int lane  = threadIdx.x & 31;
    const int mtile = blockIdx.x * 8 + wave;
    const int row16 = lane & 15;                   // A row / B col / C col
    const int g     = lane >> 4;                   // half-wave group

    long row  = (long)mtile * 16 + row16;
    long arow = row < M ? row : (M - 1);           // clamp loads for tail tiles
    const __bf16* Ar = A + arow * (long)K;

    v8f acc[NT];
#pragma unroll
    for (int t = 0; t < NT; ++t) acc[t] = (v8f){};

    for (int kc = 0; kc < K; kc += KC) {
        // stage W[kc:kc+KC, 0:N] into LDS as [n][k]
        for (int idx = threadIdx.x; idx < KC * N; idx += 256) {
            int kk = idx >> NSHIFT, nn = idx & (N - 1);
            ldsW[nn * LDSW_STRIDE + kk] = W[(long)(kc + kk) * N + nn];
        }
        __syncthreads();

        if (kc + KC < K)   // hint next A chunk toward the caches
            __builtin_prefetch(Ar + kc + KC + g * 64, 0, 1);

#pragma unroll
        for (int kb = 0; kb < KC; kb += 32) {
            // A fragment (contiguous K order, matches B): lane holds K {g*16..g*16+15}
            v8bf alo = *(const v8bf*)(Ar + kc + kb + g * 16);
            v8bf ahi = *(const v8bf*)(Ar + kc + kb + g * 16 + 8);
            v16bf a;
#pragma unroll
            for (int e = 0; e < 8; ++e) { a[e] = alo[e]; a[e + 8] = ahi[e]; }

#pragma unroll
            for (int t = 0; t < NT; ++t) {
                // B fragment: element e -> W[kc+kb + g*16 + e][t*16 + row16]
                const v8bf* wr = (const v8bf*)(ldsW + (t * 16 + row16) * LDSW_STRIDE
                                               + kb + g * 16);
                v8bf b0 = wr[0], b1 = wr[1];
                v16bf b;
#pragma unroll
                for (int e = 0; e < 8; ++e) { b[e] = b0[e]; b[e + 8] = b1[e]; }

                acc[t] = __builtin_amdgcn_wmma_f32_16x16x32_bf16(
                             false, a, false, b, (short)0, acc[t], false, false);
            }
        }
        __syncthreads();
    }

    if ((long)mtile * 16 < M) {   // M % 16 == 0, so whole tile is valid
#pragma unroll
        for (int t = 0; t < NT; ++t)
#pragma unroll
            for (int r = 0; r < 8; ++r) {
                long m = (long)mtile * 16 + g * 8 + r;   // C/D layout: M = g*8+r
                C[m * (long)N + t * 16 + row16] = acc[t][r];
            }
    }
}

// ---------------------------------------------------------------------------
// self-loop init:  agg[i,c] = t[i,c] * dinv[i]^2      grid=(nN), block=C
// ---------------------------------------------------------------------------
__global__ void k_self_init(const float* __restrict__ t, const float* __restrict__ dinv,
                            float* __restrict__ agg, int C) {
    long node = blockIdx.x;
    int  c    = threadIdx.x;
    float di  = dinv[node];
    agg[node * C + c] = t[node * C + c] * di * di;
}

// ---------------------------------------------------------------------------
// edge scatter:  agg[dst,c] += t[src,c] * dinv[src]*dinv[dst]
// grid=(nE), block=C. t (51/25 MB) is L2-resident -> gathers+atomics at L2 BW.
// ---------------------------------------------------------------------------
__global__ void k_scatter(const float* __restrict__ t, const int* __restrict__ ei,
                          const float* __restrict__ dinv, float* __restrict__ agg,
                          int nE, int C, int nN) {
    int e = blockIdx.x;
    int c = threadIdx.x;
    int s = ei[e];
    int d = ei[nE + e];
    if ((unsigned)s >= (unsigned)nN || (unsigned)d >= (unsigned)nN) return;
    float w = dinv[s] * dinv[d];
    atomicAdd(&agg[(long)d * C + c], t[(long)s * C + c] * w);
}

// ---------------------------------------------------------------------------
// h = bf16(relu(agg + b))        grid=(nN), block=C
// ---------------------------------------------------------------------------
__global__ void k_bias_relu_cvt(const float* __restrict__ agg, const float* __restrict__ b,
                                __bf16* __restrict__ h, int C) {
    long node = blockIdx.x;
    int  c    = threadIdx.x;
    h[node * C + c] = (__bf16)fmaxf(agg[node * C + c] + b[c], 0.0f);
}

// ---------------------------------------------------------------------------
// in-place log_softmax over 64 channels (+bias): one wave32 per node,
// lane handles channels {lane, lane+32}; reductions via wave shuffles.
// ---------------------------------------------------------------------------
__global__ __launch_bounds__(256) void k_logsoftmax64(float* __restrict__ out,
                                                      const float* __restrict__ b2,
                                                      int nN) {
    int  wave = threadIdx.x >> 5;
    int  lane = threadIdx.x & 31;
    long node = (long)blockIdx.x * 8 + wave;
    if (node >= nN) return;
    float* row = out + node * 64;
    float z0 = row[lane]      + b2[lane];
    float z1 = row[lane + 32] + b2[lane + 32];
    float m = fmaxf(z0, z1);
#pragma unroll
    for (int off = 16; off > 0; off >>= 1) m = fmaxf(m, __shfl_xor(m, off, 32));
    float s = __expf(z0 - m) + __expf(z1 - m);
#pragma unroll
    for (int off = 16; off > 0; off >>= 1) s += __shfl_xor(s, off, 32);
    float lse = m + __logf(s);
    row[lane]      = z0 - lse;
    row[lane + 32] = z1 - lse;
}

// ---------------------------------------------------------------------------
// host launcher
// ---------------------------------------------------------------------------
extern "C" void kernel_launch(void* const* d_in, const int* in_sizes, int n_in,
                              void* d_out, int out_size, void* d_ws, size_t ws_size,
                              hipStream_t stream) {
    const float* x  = (const float*)d_in[0];   // [N_NODES, IN_CH]
    const float* W1 = (const float*)d_in[1];   // [IN_CH, HIDDEN]
    const float* b1 = (const float*)d_in[2];   // [HIDDEN]
    const float* W2 = (const float*)d_in[3];   // [HIDDEN, OUT_CH]
    const float* b2 = (const float*)d_in[4];   // [OUT_CH]
    const int*   ei = (const int*)d_in[5];     // [2, N_EDGES] (int32 per harness)

    // workspace carve-up (256-B aligned slabs)
    char* ws = (char*)d_ws;
    size_t off = 0;
    auto carve = [&](size_t bytes) { char* p = ws + off; off += (bytes + 255) & ~(size_t)255; return p; };
    float*  dinv = (float*) carve((size_t)N_NODES * 4);                 // deg -> dinv in place
    __bf16* xb   = (__bf16*)carve((size_t)N_NODES * KPAD1 * 2);        // 102.4 MB (reused as t2)
    __bf16* w1b  = (__bf16*)carve((size_t)KPAD1 * HIDDEN * 2);
    float*  t1   = (float*) carve((size_t)N_NODES * HIDDEN * 4);       // 51.2 MB
    float*  agg1 = (float*) carve((size_t)N_NODES * HIDDEN * 4);       // 51.2 MB
    __bf16* hb   = (__bf16*)carve((size_t)N_NODES * HIDDEN * 2);       // 25.6 MB
    __bf16* w2b  = (__bf16*)carve((size_t)HIDDEN * OUT_CH * 2);
    float*  t2   = (float*)xb;                                         // reuse: xb dead after GEMM1
    float*  outp = (float*)d_out;                                      // agg2 lives in d_out

    // 1) normalization: deg (self-loop = 1) -> atomic count -> rsqrt
    k_fill<<<(N_NODES + 255) / 256, 256, 0, stream>>>(dinv, 1.0f, N_NODES);
    k_deg<<<(N_EDGES + 255) / 256, 256, 0, stream>>>(ei, dinv, N_EDGES, N_NODES);
    k_rsqrt_inplace<<<(N_NODES + 255) / 256, 256, 0, stream>>>(dinv, N_NODES);

    // 2) bf16 conversions (pad K for WMMA)
    k_cvt_pad_cols<<<dim3(KPAD1 / 256, N_NODES), 256, 0, stream>>>(x, xb, IN_CH, KPAD1);
    k_cvt_pad_rows<<<(KPAD1 * HIDDEN + 255) / 256, 256, 0, stream>>>(W1, w1b, IN_CH, KPAD1, HIDDEN);
    k_cvt_pad_rows<<<(HIDDEN * OUT_CH + 255) / 256, 256, 0, stream>>>(W2, w2b, HIDDEN, HIDDEN, OUT_CH);

    const int mblocks = (N_NODES / 16 + 7) / 8;   // 782

    // 3) layer 1: GEMM -> self-loop init -> edge scatter -> bias+relu (to bf16)
    k_gemm_wmma<HIDDEN><<<mblocks, 256, 0, stream>>>(xb, w1b, t1, N_NODES, KPAD1);
    k_self_init<<<N_NODES, HIDDEN, 0, stream>>>(t1, dinv, agg1, HIDDEN);
    k_scatter<<<N_EDGES, HIDDEN, 0, stream>>>(t1, ei, dinv, agg1, N_EDGES, HIDDEN, N_NODES);
    k_bias_relu_cvt<<<N_NODES, HIDDEN, 0, stream>>>(agg1, b1, hb, HIDDEN);

    // 4) layer 2: GEMM -> self-loop init (into d_out) -> edge scatter -> log_softmax
    k_gemm_wmma<OUT_CH><<<mblocks, 256, 0, stream>>>(hb, w2b, t2, N_NODES, HIDDEN);
    k_self_init<<<N_NODES, OUT_CH, 0, stream>>>(t2, dinv, outp, OUT_CH);
    k_scatter<<<N_EDGES, OUT_CH, 0, stream>>>(t2, ei, dinv, outp, N_EDGES, OUT_CH, N_NODES);
    k_logsoftmax64<<<(N_NODES + 7) / 8, 256, 0, stream>>>(outp, b2, N_NODES);

    (void)in_sizes; (void)n_in; (void)out_size; (void)ws_size;
}
